// NTM_41884521071297
// MI455X (gfx1250) — matrix operand order repaired and checked by reading, not compiled, not run on hardware
//
#include <hip/hip_runtime.h>
#include <math.h>

// ---------------- problem constants ----------------
#define Bsz 256
#define Tt 64
#define Hh 4
#define Nn 256
#define Mm 64
#define Ss 3
#define IN_DIM 66      // CMD+DATA
#define CIN 322        // IN_DIM + H*M
#define HID 512
#define CINS 834       // CIN + HID
#define CSP 864        // CINS padded to 27*32
#define KT1 11         // ceil(322/32)
#define KT2 27         // ceil(834/32)
#define UPDW 792       // H*(3M+S+3)
#define UPDP 800       // padded to 50*16
#define BT 16          // batch rows per workgroup (WMMA M)
#define EPSf 1e-8f

// ---------------- packed bf16 weight tiles (in d_ws) ----------------
#define NT0 32                         // HID/16
#define NT1 4                          // 64/16
#define NT2 50                         // 800/16
#define TILE_HALFS 512                 // 32x16 bf16 tile
#define W0_OFF 0
#define W0_TILES (NT0*KT1)             // 352
#define W1_OFF (W0_TILES*TILE_HALFS)
#define W1_TILES (NT1*KT2)             // 108
#define W2_OFF (W1_OFF + W1_TILES*TILE_HALFS)
#define W2_TILES (NT2*KT2)             // 1350
#define TOTAL_TILES (W0_TILES + W1_TILES + W2_TILES)  // 1810

// ---------------- d_out layout (floats) ----------------
#define OFF_OUTS 0
#define OFF_CTRL (Bsz*Tt*64)                 // 1,048,576
#define OFF_WT   (OFF_CTRL + Bsz*HID)        // 1,179,648
#define OFF_WTD  (OFF_WT + Bsz*Hh*Nn)        // 1,441,792
#define OFF_MEM  (OFF_WTD + Bsz*Hh*Nn)       // 1,703,936

typedef float  v8f   __attribute__((ext_vector_type(8)));
typedef __bf16 v16bf __attribute__((ext_vector_type(16)));

__device__ __forceinline__ unsigned short f2bf(float f){
  unsigned int u = __float_as_uint(f);
  u += 0x7fffu + ((u >> 16) & 1u);           // round-to-nearest-even
  return (unsigned short)(u >> 16);
}
__device__ __forceinline__ float frcp(float x){ return __builtin_amdgcn_rcpf(x); }   // v_rcp_f32
__device__ __forceinline__ float sigm(float x){ return frcp(1.f + __expf(-x)); }
__device__ __forceinline__ float softplusf(float x){ return x > 20.f ? x : log1pf(__expf(x)); }
__device__ __forceinline__ float wredsum(float v){
  #pragma unroll
  for (int o = 16; o > 0; o >>= 1) v += __shfl_xor(v, o, 32);
  return v;
}
__device__ __forceinline__ float wredmax(float v){
  #pragma unroll
  for (int o = 16; o > 0; o >>= 1) v = fmaxf(v, __shfl_xor(v, o, 32));
  return v;
}

// ---- pack fp32 weights -> bf16 WMMA B tiles (K x 16, per-lane contiguous) ----
__global__ void ntm_pack_weights(const float* __restrict__ Ws,
                                 const float* __restrict__ Wo,
                                 const float* __restrict__ Wu,
                                 unsigned short* __restrict__ wpk){
  int tile = blockIdx.x;
  const float* W; int Kdim, Ndim, nt, kt; size_t base;
  if (tile < W0_TILES){
    W = Ws; Kdim = CIN; Ndim = HID; nt = tile / KT1; kt = tile % KT1;
    base = (size_t)W0_OFF + (size_t)tile * TILE_HALFS;
  } else if (tile < W0_TILES + W1_TILES){
    int r = tile - W0_TILES;
    W = Wo; Kdim = CINS; Ndim = 64; nt = r / KT2; kt = r % KT2;
    base = (size_t)W1_OFF + (size_t)r * TILE_HALFS;
  } else {
    int r = tile - W0_TILES - W1_TILES;
    W = Wu; Kdim = CINS; Ndim = UPDW; nt = r / KT2; kt = r % KT2;
    base = (size_t)W2_OFF + (size_t)r * TILE_HALFS;
  }
  for (int e = threadIdx.x; e < TILE_HALFS; e += blockDim.x){
    int ln = e >> 4, i = e & 15;
    // B layout: lanes 0-15 -> K 0..7,16..23, lanes 16-31 -> K 8..15,24..31; col = lane&15
    int k = kt*32 + ((ln >= 16) ? 8 : 0) + (i < 8 ? i : i + 8);
    int n = nt*16 + (ln & 15);
    float v = (k < Kdim && n < Ndim) ? W[(size_t)k * Ndim + n] : 0.f;
    wpk[base + e] = f2bf(v);
  }
}

__global__ void ntm_copy_wtd(const float* __restrict__ src, float* __restrict__ dst, int n){
  int i = blockIdx.x * blockDim.x + threadIdx.x;
  if (i < n) dst[i] = src[i];
}

// ---- persistent NTM scan: one WG owns 16 batch rows for all 64 steps ----
__global__ __launch_bounds__(256, 1)
void ntm_scan(const float* __restrict__ x,
              const float* __restrict__ wt_in,
              const float* __restrict__ mem_in,
              const float* __restrict__ b_i2s,
              const float* __restrict__ b_i2o,
              const float* __restrict__ b_i2u,
              const unsigned short* __restrict__ wpk,
              float* __restrict__ out){
  extern __shared__ char smem[];
  unsigned short* cs   = (unsigned short*)(smem);            // [16][864] bf16 activations  27648 B
  unsigned short* pA   = (unsigned short*)(smem + 27648);    // [27][512] packed A tiles    27648 B
  float* updL  = (float*)(smem + 55296);                     // [16][800] upd               51200 B
  float* wtL   = (float*)(smem + 106496);                    // [16][4][256] wt state       65536 B
  float* W2L   = (float*)(smem + 172032);                    // [16][4][256] scratch        65536 B
  float* knL   = (float*)(smem + 237568);                    // [16][4][64] normed keys     16384 B
  float* betaL = (float*)(smem + 253952);                    // [64]
  float* gL    = (float*)(smem + 254208);                    // [64]
  float* gammaL= (float*)(smem + 254464);                    // [64]
  float* sL    = (float*)(smem + 254720);                    // [64][3]   (total 255488 B)

  const int tid  = threadIdx.x;
  const int lane = tid & 31;
  const int wave = tid >> 5;
  const int b0   = blockIdx.x * BT;

  float* outs   = out + OFF_OUTS;
  float* outCtr = out + OFF_CTRL;
  float* outWt  = out + OFF_WT;
  float* memS   = out + OFF_MEM;      // working + final memory state

  // ---- init persistent state ----
  for (int j = tid; j < BT*Hh*Nn; j += 256) wtL[j] = wt_in[(size_t)b0*Hh*Nn + j];
  {
    const float4* src = (const float4*)(mem_in + (size_t)b0*Nn*Mm);
    float4*       dst = (float4*)(memS + (size_t)b0*Nn*Mm);
    for (int j = tid; j < BT*Nn*Mm/4; j += 256) dst[j] = src[j];
  }
  for (int j = tid; j < BT*(CSP-CINS); j += 256){             // zero pad cols [834,864)
    int b = j / (CSP-CINS), c = CINS + j % (CSP-CINS);
    cs[b*CSP + c] = 0;
  }
  __threadfence();
  __syncthreads();

  for (int t = 0; t < Tt; ++t){
    // ---- A: read = wt @ mem ; comb -> cs (bf16) ----
    for (int j = tid; j < BT*IN_DIM; j += 256){
      int b = j / IN_DIM, c = j % IN_DIM;
      cs[b*CSP + c] = f2bf(x[((size_t)(b0+b)*Tt + t)*IN_DIM + c]);
    }
    for (int j = tid; j < BT*30; j += 256){                   // zero [322,352) for GEMM1 tail tile
      int b = j / 30, c = CIN + (j % 30);
      cs[b*CSP + c] = 0;
    }
    for (int idx = tid; idx < BT*Hh*Mm; idx += 256){
      int b = idx >> 8, rem = idx & 255;
      int h = rem >> 6, m = rem & 63;
      const float* mrow = memS + (size_t)(b0+b)*Nn*Mm + m;
      const float* wrow = wtL + (b*Hh + h)*Nn;
      float acc = 0.f;
      for (int n = 0; n < Nn; ++n) acc += wrow[n] * mrow[(size_t)n*Mm];
      cs[b*CSP + IN_DIM + h*Mm + m] = f2bf(acc);
    }
    __syncthreads();

    // ---- B: pack A tiles 0..10 (16x32 bf16, per-lane contiguous) ----
    for (int e = tid; e < KT1*TILE_HALFS; e += 256){
      int tl = e >> 9, r = e & 511;
      int ln = r >> 4, i = r & 15;
      int row = ln & 15;
      int K = tl*32 + ((ln >= 16) ? 8 : 0) + (i < 8 ? i : i + 8);
      pA[tl*TILE_HALFS + r] = cs[row*CSP + K];
    }
    __syncthreads();

    // ---- C: GEMM1 (16x322)@(322x512) -> ctrl = sigmoid(...) ----
    for (int nt = wave; nt < NT0; nt += 8){
      float bv = b_i2s[nt*16 + (lane & 15)];
      v8f acc;
      #pragma unroll
      for (int j = 0; j < 8; ++j) acc[j] = bv;
      for (int kt = 0; kt < KT1; ++kt){
        v16bf a  = *reinterpret_cast<const v16bf*>(&pA[kt*TILE_HALFS + lane*16]);
        v16bf bm = *reinterpret_cast<const v16bf*>(&wpk[(size_t)W0_OFF + (size_t)(nt*KT1 + kt)*TILE_HALFS + lane*16]);
        acc = __builtin_amdgcn_wmma_f32_16x16x32_bf16(false, a, false, bm, (short)0, acc, false, false);
      }
      int rbase = (lane >= 16) ? 8 : 0;
      int col   = lane & 15;
      #pragma unroll
      for (int j = 0; j < 8; ++j){
        int row = j + rbase;
        float v = sigm(acc[j]);
        int n = nt*16 + col;
        cs[row*CSP + CIN + n] = f2bf(v);
        if (t == Tt-1) outCtr[(size_t)(b0+row)*HID + n] = v;
      }
    }
    __syncthreads();

    // ---- D: pack A tiles 10..26 (now include ctrl part of cs) ----
    for (int e = tid; e < (KT2-10)*TILE_HALFS; e += 256){
      int tl = 10 + (e >> 9), r = e & 511;
      int ln = r >> 4, i = r & 15;
      int row = ln & 15;
      int K = tl*32 + ((ln >= 16) ? 8 : 0) + (i < 8 ? i : i + 8);
      pA[tl*TILE_HALFS + r] = cs[row*CSP + K];
    }
    __syncthreads();

    // ---- E: GEMM2/3 (16x834)@(834x64 | 834x792) -> out_t, upd ----
    for (int nt = wave; nt < NT1 + NT2; nt += 8){
      float bv; size_t wbase;
      if (nt < NT1){
        bv = b_i2o[nt*16 + (lane & 15)];
        wbase = (size_t)W1_OFF + (size_t)(nt*KT2)*TILE_HALFS;
      } else {
        int n2 = (nt-NT1)*16 + (lane & 15);
        bv = (n2 < UPDW) ? b_i2u[n2] : 0.f;
        wbase = (size_t)W2_OFF + (size_t)((nt-NT1)*KT2)*TILE_HALFS;
      }
      v8f acc;
      #pragma unroll
      for (int j = 0; j < 8; ++j) acc[j] = bv;
      for (int kt = 0; kt < KT2; ++kt){
        v16bf a  = *reinterpret_cast<const v16bf*>(&pA[kt*TILE_HALFS + lane*16]);
        v16bf bm = *reinterpret_cast<const v16bf*>(&wpk[wbase + (size_t)kt*TILE_HALFS + lane*16]);
        acc = __builtin_amdgcn_wmma_f32_16x16x32_bf16(false, a, false, bm, (short)0, acc, false, false);
      }
      int rbase = (lane >= 16) ? 8 : 0;
      int col   = lane & 15;
      #pragma unroll
      for (int j = 0; j < 8; ++j){
        int row = j + rbase;
        if (nt < NT1){
          outs[((size_t)(b0+row)*Tt + t)*64 + nt*16 + col] = sigm(acc[j]);
        } else {
          updL[row*UPDP + (nt-NT1)*16 + col] = acc[j];
        }
      }
    }
    __syncthreads();

    // ---- F: per-head scalars + normalized keys; sigmoid(erase)/tanh(add) in place ----
    for (int e = tid; e < BT*Hh*2*Mm; e += 256){
      int b = e >> 9, rem = e & 511;
      int h = rem >> 7, q = rem & 127;
      int col = h*198 + 70 + q;                 // erase [70,134), add [134,198)
      float v = updL[b*UPDP + col];
      updL[b*UPDP + col] = (q < Mm) ? sigm(v) : tanhf(v);
    }
    if (tid < BT*Hh){
      int b = tid >> 2, h = tid & 3;
      float* ur = updL + b*UPDP + h*198;
      float nrm = 0.f;
      for (int m = 0; m < Mm; ++m){ float k = ur[m]; nrm += k*k; }
      float inv = frcp(sqrtf(nrm) + EPSf);
      float* kn = knL + tid*Mm;
      for (int m = 0; m < Mm; ++m) kn[m] = ur[m]*inv;
      betaL[tid] = softplusf(ur[Mm]);
      gL[tid]    = sigm(ur[Mm+1]);
      float e0 = ur[Mm+2], e1 = ur[Mm+3], e2 = ur[Mm+4];
      float mx = fmaxf(e0, fmaxf(e1, e2));
      float x0 = __expf(e0-mx), x1 = __expf(e1-mx), x2 = __expf(e2-mx);
      float si = frcp(x0+x1+x2);
      sL[tid*3+0] = x0*si; sL[tid*3+1] = x1*si; sL[tid*3+2] = x2*si;
      gammaL[tid] = 1.f + softplusf(ur[Mm+2+Ss]);
    }
    __syncthreads();

    // ---- G: cosine content scores -> W2L ----
    for (int idx = tid; idx < BT*Nn; idx += 256){
      int b = idx >> 8, n = idx & 255;
      const float* mrow = memS + (size_t)(b0+b)*Nn*Mm + (size_t)n*Mm;
      const float* k0 = knL + (b*Hh+0)*Mm;
      const float* k1 = knL + (b*Hh+1)*Mm;
      const float* k2 = knL + (b*Hh+2)*Mm;
      const float* k3 = knL + (b*Hh+3)*Mm;
      float nrm = 0.f, d0 = 0.f, d1 = 0.f, d2 = 0.f, d3 = 0.f;
      for (int m = 0; m < Mm; ++m){
        float v = mrow[m];
        nrm += v*v;
        d0 += k0[m]*v; d1 += k1[m]*v; d2 += k2[m]*v; d3 += k3[m]*v;
      }
      float inv = frcp(sqrtf(nrm) + EPSf);
      W2L[(b*Hh+0)*Nn + n] = betaL[b*Hh+0]*d0*inv;
      W2L[(b*Hh+1)*Nn + n] = betaL[b*Hh+1]*d1*inv;
      W2L[(b*Hh+2)*Nn + n] = betaL[b*Hh+2]*d2*inv;
      W2L[(b*Hh+3)*Nn + n] = betaL[b*Hh+3]*d3*inv;
    }
    __syncthreads();

    // ---- H: softmax over N (wave32 reductions) + interpolation gate ----
    for (int rr = 0; rr < 8; ++rr){
      int row = wave*8 + rr;                   // row = b*H + h
      float* wc = W2L + row*Nn;
      const float* wo = wtL + row*Nn;
      float g = gL[row];
      float v[8]; float mx = -3.4e38f;
      #pragma unroll
      for (int jj = 0; jj < 8; ++jj){ v[jj] = wc[jj*32 + lane]; mx = fmaxf(mx, v[jj]); }
      mx = wredmax(mx);
      float sum = 0.f;
      #pragma unroll
      for (int jj = 0; jj < 8; ++jj){ v[jj] = __expf(v[jj]-mx); sum += v[jj]; }
      sum = wredsum(sum);
      float inv = frcp(sum);
      #pragma unroll
      for (int jj = 0; jj < 8; ++jj){
        int n = jj*32 + lane;
        wc[n] = g*(v[jj]*inv) + (1.f-g)*wo[n];
      }
    }
    __syncthreads();

    // ---- I: circular shift + sharpen + renormalize -> new wt ----
    for (int rr = 0; rr < 8; ++rr){
      int row = wave*8 + rr;
      const float* wg = W2L + row*Nn;
      float* wn = wtL + row*Nn;
      float s0 = sL[row*3+0], s1 = sL[row*3+1], s2 = sL[row*3+2];
      float gam = gammaL[row];
      float p[8]; float sum = 0.f;
      #pragma unroll
      for (int jj = 0; jj < 8; ++jj){
        int n = jj*32 + lane;
        float ws = s0*wg[(n+1)&255] + s1*wg[n] + s2*wg[(n-1)&255];
        float v = __powf(ws + EPSf, gam);
        p[jj] = v; sum += v;
      }
      sum = wredsum(sum);
      float inv = frcp(sum);
      #pragma unroll
      for (int jj = 0; jj < 8; ++jj) wn[jj*32 + lane] = p[jj]*inv;
    }
    __syncthreads();

    // ---- J: memory erase/add update (in place in memS) ----
    for (int idx = tid; idx < BT*Nn; idx += 256){
      int b = idx >> 8, n = idx & 255;
      float* mrow = memS + (size_t)(b0+b)*Nn*Mm + (size_t)n*Mm;
      float w0 = wtL[(b*Hh+0)*Nn + n];
      float w1 = wtL[(b*Hh+1)*Nn + n];
      float w2 = wtL[(b*Hh+2)*Nn + n];
      float w3 = wtL[(b*Hh+3)*Nn + n];
      const float* u = updL + b*UPDP;
      for (int m = 0; m < Mm; ++m){
        float keep = (1.f - w0*u[0*198+70+m]) * (1.f - w1*u[1*198+70+m])
                   * (1.f - w2*u[2*198+70+m]) * (1.f - w3*u[3*198+70+m]);
        float addv = w0*u[0*198+134+m] + w1*u[1*198+134+m]
                   + w2*u[2*198+134+m] + w3*u[3*198+134+m];
        mrow[m] = mrow[m]*keep + addv;
      }
    }
    __threadfence();
    __syncthreads();
  }

  // ---- finals: wt_f ----
  for (int j = tid; j < BT*Hh*Nn; j += 256) outWt[(size_t)b0*Hh*Nn + j] = wtL[j];
}

extern "C" void kernel_launch(void* const* d_in, const int* in_sizes, int n_in,
                              void* d_out, int out_size, void* d_ws, size_t ws_size,
                              hipStream_t stream){
  const float* x   = (const float*)d_in[0];
  const float* wt  = (const float*)d_in[2];
  const float* wtd = (const float*)d_in[3];
  const float* mem = (const float*)d_in[4];
  const float* Ws  = (const float*)d_in[5];
  const float* bs  = (const float*)d_in[6];
  const float* Wo  = (const float*)d_in[7];
  const float* bo  = (const float*)d_in[8];
  const float* Wu  = (const float*)d_in[9];
  const float* bu  = (const float*)d_in[10];
  float* out = (float*)d_out;
  unsigned short* wpk = (unsigned short*)d_ws;   // ~1.81 MB packed bf16 weights

  ntm_pack_weights<<<TOTAL_TILES, 256, 0, stream>>>(Ws, Wo, Wu, wpk);
  ntm_copy_wtd<<<(Bsz*Hh*Nn + 255)/256, 256, 0, stream>>>(wtd, out + OFF_WTD, Bsz*Hh*Nn);
  ntm_scan<<<Bsz/BT, 256, 255488, stream>>>(x, wt, mem, bs, bo, bu, wpk, out);
}
